// AdvancedHybridModel_74182675137132
// MI455X (gfx1250) — compile-verified, matrix operand order checked by reading, not run on hardware
//
#include <hip/hip_runtime.h>

// ---------------------------------------------------------------------------
// AdvancedHybridModel forward for MI455X (gfx1250, wave32, WMMA)
//   - classical GEMMs: bf16 WMMA (v_wmma_f32_16x16x32_bf16), fp32 epilogues,
//     4 waves/block, unconditional global_prefetch 4 K-tiles ahead
//   - 16-qubit state-vector sim: fused 2x2 gates, 32KB LDS chunk pass for the
//     12 low bit positions (loaded/stored with GLOBAL_*_ASYNC_*_LDS_B128 on
//     the ASYNCcnt path), global pair passes for the 4 high ones, in-place
//     CNOT swap passes, fused per-layer CZ sign pass, single-pass expvals.
// ---------------------------------------------------------------------------

#define NBATCH 128
#define NQ     16
#define NST    65536            // 2^16 amplitudes
#define NLAY   8
#define PI_F   3.14159265358979f

typedef __attribute__((ext_vector_type(16))) __bf16         bfx16;
typedef __attribute__((ext_vector_type(16))) unsigned short usx16;
typedef __attribute__((ext_vector_type(8)))  float          fx8;

// ---------------- workspace layout (bytes) ----------------
constexpr size_t AL(size_t x) { return (x + 255) & ~(size_t)255; }
constexpr size_t OFF_STATE = 0;
constexpr size_t OFF_ENCG  = AL(OFF_STATE + (size_t)NBATCH * NST * 8);   // 64MB state
constexpr size_t OFF_LAYG  = AL(OFF_ENCG  + (size_t)NBATCH * 16 * 8 * 4);
constexpr size_t OFF_C     = AL(OFF_LAYG  + (size_t)NLAY * 16 * 8 * 4);
constexpr size_t OFF_SK1   = AL(OFF_C     + 128 * 512 * 4);
constexpr size_t OFF_SK2   = AL(OFF_SK1   + 128 * 256 * 4);
constexpr size_t OFF_XPRE  = AL(OFF_SK2   + 128 * 128 * 4);
constexpr size_t OFF_QOUT  = AL(OFF_XPRE  + 128 * 16 * 4);
constexpr size_t OFF_BF    = AL(OFF_QOUT  + 128 * 52 * 4);
// bf16 (ushort) staging arena: weights (padded) + activations
constexpr size_t OFF_XS   = OFF_BF;
constexpr size_t OFF_W1S  = AL(OFF_XS   + 128*512*2);
constexpr size_t OFF_W2S  = AL(OFF_W1S  + 512*512*2);
constexpr size_t OFF_W3S  = AL(OFF_W2S  + 256*512*2);
constexpr size_t OFF_WP1S = AL(OFF_W3S  + 128*256*2);
constexpr size_t OFF_WP2S = AL(OFF_WP1S + 64*128*2);
constexpr size_t OFF_WQ1S = AL(OFF_WP2S + 16*64*2);
constexpr size_t OFF_WQ2S = AL(OFF_WQ1S + 256*64*2);
constexpr size_t OFF_WO1S = AL(OFF_WQ2S + 128*256*2);
constexpr size_t OFF_WO2S = AL(OFF_WO1S + 64*128*2);
constexpr size_t OFF_WO3S = AL(OFF_WO2S + 32*64*2);
constexpr size_t OFF_WS1S = AL(OFF_WO3S + 16*32*2);
constexpr size_t OFF_WS2S = AL(OFF_WS1S + 256*512*2);
constexpr size_t OFF_A1S  = AL(OFF_WS2S + 128*512*2);
constexpr size_t OFF_A2S  = AL(OFF_A1S  + 128*512*2);
constexpr size_t OFF_A3S  = AL(OFF_A2S  + 128*256*2);
constexpr size_t OFF_ATS  = AL(OFF_A3S  + 128*128*2);
constexpr size_t OFF_AQS  = AL(OFF_ATS  + 128*64*2);
constexpr size_t OFF_AP1S = AL(OFF_AQS  + 128*64*2);
constexpr size_t OFF_AP2S = AL(OFF_AP1S + 128*256*2);
constexpr size_t OFF_AH1S = AL(OFF_AP2S + 128*128*2);
constexpr size_t OFF_AH2S = AL(OFF_AH1S + 128*64*2);
constexpr size_t OFF_END  = AL(OFF_AH2S + 128*32*2);

// ---------------- small device helpers ----------------
__device__ __forceinline__ unsigned short f2bf(float f) {
    unsigned u = __float_as_uint(f);
    u += 0x7FFFu + ((u >> 16) & 1u);          // round-to-nearest-even
    return (unsigned short)(u >> 16);
}
__device__ __forceinline__ float silu_f(float v) { return v / (1.f + __expf(-v)); }

__device__ __forceinline__ float2 cml(float2 p, float2 q) {
    return make_float2(p.x*q.x - p.y*q.y, p.x*q.y + p.y*q.x);
}
__device__ __forceinline__ float2 cad(float2 p, float2 q) {
    return make_float2(p.x + q.x, p.y + q.y);
}
struct cM { float2 a, b, c, d; };             // [[a,b],[c,d]]
__device__ __forceinline__ cM mm2(const cM& A, const cM& B) {
    cM R;
    R.a = cad(cml(A.a, B.a), cml(A.b, B.c));
    R.b = cad(cml(A.a, B.b), cml(A.b, B.d));
    R.c = cad(cml(A.c, B.a), cml(A.d, B.c));
    R.d = cad(cml(A.c, B.b), cml(A.d, B.d));
    return R;
}
__device__ __forceinline__ void storeG(float* p, const cM& F) {
    p[0]=F.a.x; p[1]=F.a.y; p[2]=F.b.x; p[3]=F.b.y;
    p[4]=F.c.x; p[5]=F.c.y; p[6]=F.d.x; p[7]=F.d.y;
}

// ---------------- utility kernels ----------------
__global__ void zero_f32_kernel(float* p, int n) {
    for (int i = blockIdx.x * blockDim.x + threadIdx.x; i < n; i += gridDim.x * blockDim.x)
        p[i] = 0.f;
}

__global__ void cvt_pad_kernel(const float* __restrict__ src, unsigned short* __restrict__ dst,
                               int rows, int cols, int colsPad, int total) {
    for (int i = blockIdx.x * blockDim.x + threadIdx.x; i < total; i += gridDim.x * blockDim.x) {
        int r = i / colsPad, c = i - r * colsPad;
        float v = (r < rows && c < cols) ? src[r * cols + c] : 0.f;
        dst[i] = f2bf(v);
    }
}

// ---------------- WMMA GEMM: C[128xN] = A[128xK] * B[NxK]^T ----------------
// A: bf16 staged, ld = K (K multiple of 32, zero padded)
// B: bf16 staged, Npad x K (Npad multiple of 16, zero padded)
// 128-thread block = 4 waves; each wave computes one 16x16 output tile.
// Prefetch is unconditional (fixed distance): keeps the inner loop straight-
// line so the scheduler pipelines the fragment loads across WMMA steps;
// over-prefetch past the staging buffer is benign (no data, no error, ISA 10.5).
__global__ void gemm_wmma_kernel(const unsigned short* __restrict__ A,
                                 const unsigned short* __restrict__ B,
                                 float* __restrict__ C, int N, int K) {
    const int wave = threadIdx.x >> 5;
    const int lane = threadIdx.x & 31;
    const int n0   = (blockIdx.x * 4 + wave) * 16;
    const int m0   = blockIdx.y * 16;
    const int Npad = (N + 15) & ~15;
    if (n0 >= Npad) return;                   // uniform per wave, no barriers here
    const int half = lane >> 4;               // K-block selector (ISA A-frag layout)
    const int r    = lane & 15;               // row (A) / col (B)

    fx8 acc = {0.f,0.f,0.f,0.f,0.f,0.f,0.f,0.f};
    const unsigned short* ap = A + (size_t)(m0 + r) * K + half * 8;
    const unsigned short* bp = B + (size_t)(n0 + r) * K + half * 8;

    for (int k = 0; k < K; k += 32) {
        __builtin_prefetch(ap + 128, 0, 3);   // 4 K-tiles (256B) ahead
        __builtin_prefetch(bp + 128, 0, 3);
        usx16 ua, ub;
#pragma unroll
        for (int j = 0; j < 8; ++j) {
            ua[j]     = ap[j];    ub[j]     = bp[j];      // K = half*8 + j
            ua[j + 8] = ap[16+j]; ub[j + 8] = bp[16+j];   // K = half*8 + 16 + j
        }
        ap += 32; bp += 32;
        acc = __builtin_amdgcn_wmma_f32_16x16x32_bf16(
                false, __builtin_bit_cast(bfx16, ua),
                false, __builtin_bit_cast(bfx16, ub),
                (short)0, acc, false, false);
    }
    if (n0 + r < N) {
#pragma unroll
        for (int i = 0; i < 8; ++i)
            C[(size_t)(m0 + i + half * 8) * N + n0 + r] = acc[i];
    }
}

// ---------------- epilogue: bias [+LN] [+act] [+skip], fp32/bf16 out --------
// mode: 0=bias, 1=bias+LN+SiLU, 2=bias+SiLU, 3=bias+tanh. skip added last.
__global__ void epilogue_kernel(const float* __restrict__ C, const float* __restrict__ bias,
                                const float* __restrict__ skip,
                                const float* __restrict__ gamma, const float* __restrict__ beta,
                                float* __restrict__ outF, unsigned short* __restrict__ outB,
                                int N, int ldb, int mode) {
    const int m = blockIdx.x, tid = threadIdx.x;
    __shared__ float r1[256], r2[256];
    float mean = 0.f, rstd = 1.f;
    if (mode == 1) {
        float s1 = 0.f, s2 = 0.f;
        for (int n = tid; n < N; n += 256) {
            float v = C[m * N + n] + bias[n];
            s1 += v; s2 += v * v;
        }
        r1[tid] = s1; r2[tid] = s2; __syncthreads();
        for (int s = 128; s > 0; s >>= 1) {
            if (tid < s) { r1[tid] += r1[tid + s]; r2[tid] += r2[tid + s]; }
            __syncthreads();
        }
        mean = r1[0] / N;
        float var = r2[0] / N - mean * mean;
        rstd = rsqrtf(var + 1e-5f);
    }
    for (int n = tid; n < N; n += 256) {
        float v = C[m * N + n] + bias[n];
        if      (mode == 1) { v = (v - mean) * rstd * gamma[n] + beta[n]; v = silu_f(v); }
        else if (mode == 2) { v = silu_f(v); }
        else if (mode == 3) { v = tanhf(v); }
        if (skip) v += skip[m * N + n];
        if (outF) outF[m * N + n] = v;
        if (outB) outB[(size_t)m * ldb + n] = f2bf(v);
    }
}

// ---------------- quantum: fused-gate preparation ----------------
// encoding: per (batch,qubit): F = H * RZ(a) * RY(a), a = xpre*pi
__global__ void prep_enc_kernel(const float* __restrict__ xpre, float* __restrict__ encg) {
    int id = blockIdx.x * blockDim.x + threadIdx.x;
    if (id >= NBATCH * NQ) return;
    float a = xpre[id] * PI_F;
    float sh, ch; __sincosf(0.5f * a, &sh, &ch);
    cM RY = { make_float2(ch,0), make_float2(-sh,0), make_float2(sh,0), make_float2(ch,0) };
    cM RZ = { make_float2(ch,-sh), make_float2(0,0), make_float2(0,0), make_float2(ch,sh) };
    const float h = 0.70710678118654752f;
    cM H  = { make_float2(h,0), make_float2(h,0), make_float2(h,0), make_float2(-h,0) };
    cM F = mm2(H, mm2(RZ, RY));
    storeG(encg + (size_t)id * 8, F);
}

// layer: per (layer,qubit): F = U3(w0,w1,w2) * RZ(w2) * RY(w1) * RX(w0)
__global__ void prep_layer_kernel(const float* __restrict__ qw, float* __restrict__ layg) {
    int id = threadIdx.x;                      // 0..127 = l*16+i
    if (id >= NLAY * NQ) return;
    float w0 = qw[id * 3 + 0], w1 = qw[id * 3 + 1], w2 = qw[id * 3 + 2];
    float s0, c0; __sincosf(0.5f * w0, &s0, &c0);
    float s1, c1; __sincosf(0.5f * w1, &s1, &c1);
    float s2, c2; __sincosf(0.5f * w2, &s2, &c2);
    cM RX = { make_float2(c0,0), make_float2(0,-s0), make_float2(0,-s0), make_float2(c0,0) };
    cM RY = { make_float2(c1,0), make_float2(-s1,0), make_float2(s1,0), make_float2(c1,0) };
    cM RZ = { make_float2(c2,-s2), make_float2(0,0), make_float2(0,0), make_float2(c2,s2) };
    float cp, sp; __sincosf(w1, &sp, &cp);           // e^{i phi}
    float cl, sl; __sincosf(w2, &sl, &cl);           // e^{i lambda}
    float cpl, spl; __sincosf(w1 + w2, &spl, &cpl);  // e^{i (phi+lambda)}
    cM U3 = { make_float2(c0, 0),
              make_float2(-cl * s0, -sl * s0),
              make_float2(cp * s0,  sp * s0),
              make_float2(cpl * c0, spl * c0) };
    cM F = mm2(U3, mm2(RZ, mm2(RY, RX)));
    storeG(layg + (size_t)id * 8, F);
}

// ---------------- quantum: state kernels ----------------
__global__ void q_init_kernel(float2* __restrict__ st) {
    for (size_t i = (size_t)blockIdx.x * blockDim.x + threadIdx.x;
         i < (size_t)NBATCH * NST; i += (size_t)gridDim.x * blockDim.x)
        st[i] = make_float2(((i & (NST - 1)) == 0) ? 1.f : 0.f, 0.f);
}

// LDS chunk pass: applies fused 1q gates on bit positions 0..11 (qubits 15..4).
// grid = (16 chunks, 128 batch), 256 threads, 32KB LDS.
// Chunk is moved global<->LDS with the gfx1250 async-LDS data path (ASYNCcnt):
// 16B per lane per issue, no VGPR round trip. The single static __shared__
// allocation starts at LDS offset 0, so the LDS address operand is the byte
// offset into the chunk.
__global__ void q_apply_low_kernel(float2* __restrict__ st, const float* __restrict__ gates,
                                   int batStrideFloats) {
    __shared__ float2 lds[4096];
    const int tid = threadIdx.x;
    const int bat = blockIdx.y;
    const size_t base = ((size_t)bat << 16) + ((size_t)blockIdx.x << 12);
#if defined(__gfx1250__)
    for (int i = tid * 2; i < 4096; i += 512) {          // 2 float2 = 16B per lane
        unsigned ldsoff = (unsigned)(i * 8);
        const float2* gp = st + base + i;
        asm volatile("global_load_async_to_lds_b128 %0, %1, off"
                     :: "v"(ldsoff), "v"(gp) : "memory");
    }
    asm volatile("s_wait_asynccnt 0x0" ::: "memory");
#else
    for (int i = tid; i < 4096; i += 256) lds[i] = st[base + i];
#endif
    const float* gb = gates + (size_t)bat * batStrideFloats;
    for (int bp = 0; bp < 12; ++bp) {
        __syncthreads();
        const float* g = gb + (15 - bp) * 8;       // qubit index = 15 - bitpos
        float g0=g[0],g1=g[1],g2=g[2],g3=g[3],g4=g[4],g5=g[5],g6=g[6],g7=g[7];
        unsigned mask = (1u << bp) - 1u;
        for (int p = tid; p < 2048; p += 256) {
            unsigned i0 = ((p & ~mask) << 1) | (p & mask);
            unsigned i1 = i0 | (1u << bp);
            float2 a0 = lds[i0], a1 = lds[i1];
            float2 n0, n1;
            n0.x = g0*a0.x - g1*a0.y + g2*a1.x - g3*a1.y;
            n0.y = g0*a0.y + g1*a0.x + g2*a1.y + g3*a1.x;
            n1.x = g4*a0.x - g5*a0.y + g6*a1.x - g7*a1.y;
            n1.y = g4*a0.y + g5*a0.x + g6*a1.y + g7*a1.x;
            lds[i0] = n0; lds[i1] = n1;
        }
    }
    __syncthreads();
#if defined(__gfx1250__)
    for (int i = tid * 2; i < 4096; i += 512) {
        unsigned ldsoff = (unsigned)(i * 8);
        float2* gp = st + base + i;
        asm volatile("global_store_async_from_lds_b128 %0, %1, off"
                     :: "v"(gp), "v"(ldsoff) : "memory");
    }
    asm volatile("s_wait_asynccnt 0x0" ::: "memory");
#else
    for (int i = tid; i < 4096; i += 256) st[base + i] = lds[i];
#endif
}

// global pair pass for one high bit position. grid = (128, 128), 256 threads.
__global__ void q_gate_hi_kernel(float2* __restrict__ st, const float* __restrict__ gates,
                                 int batStrideFloats, int gateIdx, int bitpos) {
    const int bat = blockIdx.y;
    const float* g = gates + (size_t)bat * batStrideFloats + (size_t)gateIdx * 8;
    float g0=g[0],g1=g[1],g2=g[2],g3=g[3],g4=g[4],g5=g[5],g6=g[6],g7=g[7];
    unsigned p = blockIdx.x * 256 + threadIdx.x;           // 0..32767
    unsigned mask = (1u << bitpos) - 1u;
    unsigned i0 = ((p & ~mask) << 1) | (p & mask);
    unsigned i1 = i0 | (1u << bitpos);
    size_t base = (size_t)bat << 16;
    float2 a0 = st[base + i0], a1 = st[base + i1];
    float2 n0, n1;
    n0.x = g0*a0.x - g1*a0.y + g2*a1.x - g3*a1.y;
    n0.y = g0*a0.y + g1*a0.x + g2*a1.y + g3*a1.x;
    n1.x = g4*a0.x - g5*a0.y + g6*a1.x - g7*a1.y;
    n1.y = g4*a0.y + g5*a0.x + g6*a1.y + g7*a1.x;
    st[base + i0] = n0; st[base + i1] = n1;
}

// in-place CNOT: swap amplitudes (ctrl=1,tgt=0) <-> (ctrl=1,tgt=1).
// bc/bt are bit positions. grid = (64, 128), 256 threads (16384 pairs).
__global__ void q_cnot_kernel(float2* __restrict__ st, int bc, int bt) {
    const int bat = blockIdx.y;
    unsigned p = blockIdx.x * 256 + threadIdx.x;           // 0..16383
    int lo = bc < bt ? bc : bt, hi = bc < bt ? bt : bc;
    unsigned ml = (1u << lo) - 1u, mh = (1u << hi) - 1u;
    unsigned y = ((p & ~ml) << 1) | (p & ml);
    unsigned z = ((y & ~mh) << 1) | (y & mh);
    unsigned iA = z | (1u << bc);
    unsigned iB = iA | (1u << bt);
    size_t base = (size_t)bat << 16;
    float2 t0 = st[base + iA], t1 = st[base + iB];
    st[base + iA] = t1; st[base + iB] = t0;
}

// fused CZ block of a layer: flip sign if parity of b_i&b_{i+1} (i even) is odd.
__global__ void q_cz_layer_kernel(float2* __restrict__ st) {
    const int bat = blockIdx.y;
    unsigned idx = blockIdx.x * 256 + threadIdx.x;         // 0..65535
    unsigned s = 0;
#pragma unroll
    for (int i = 0; i < 16; i += 2) {
        int b1 = 15 - i, b2 = 14 - i;
        s ^= ((idx >> b1) & (idx >> b2)) & 1u;
    }
    if (s) {
        size_t a = ((size_t)bat << 16) + idx;
        float2 v = st[a];
        st[a] = make_float2(-v.x, -v.y);
    }
}

// expectation values for qubit qb: Z, X, Y (+ ZZ_{4qb,4qb+1} when qb<4).
// grid = (32, 128), 256 threads; atomic accumulate into q[bat*52 + slot].
__global__ void q_expect_kernel(const float2* __restrict__ st, float* __restrict__ q, int qb) {
    const int bat = blockIdx.y;
    const int bp = 15 - qb;
    __shared__ float rz[256], rx[256], ry[256], rzz[256];
    float sz = 0.f, sx = 0.f, sy = 0.f, szz = 0.f;
    unsigned mask = (1u << bp) - 1u;
    size_t base = (size_t)bat << 16;
    int zb1 = 15 - 4 * qb, zb2 = 14 - 4 * qb;
    for (unsigned p = blockIdx.x * 256 + threadIdx.x; p < 32768; p += 32 * 256) {
        unsigned i0 = ((p & ~mask) << 1) | (p & mask);
        unsigned i1 = i0 | (1u << bp);
        float2 a0 = st[base + i0], a1 = st[base + i1];
        float n0 = a0.x*a0.x + a0.y*a0.y, n1 = a1.x*a1.x + a1.y*a1.y;
        sz += n0 - n1;
        float wr = a0.x*a1.x + a0.y*a1.y;   // Re(conj(a0)*a1)
        float wi = a0.x*a1.y - a0.y*a1.x;   // Im(conj(a0)*a1)
        sx += 2.f * wr; sy += 2.f * wi;
        if (qb < 4) {
            unsigned s0 = ((i0 >> zb1) ^ (i0 >> zb2)) & 1u;
            unsigned s1 = ((i1 >> zb1) ^ (i1 >> zb2)) & 1u;
            szz += s0 ? -n0 : n0;
            szz += s1 ? -n1 : n1;
        }
    }
    int tid = threadIdx.x;
    rz[tid] = sz; rx[tid] = sx; ry[tid] = sy; rzz[tid] = szz;
    __syncthreads();
    for (int s = 128; s > 0; s >>= 1) {
        if (tid < s) {
            rz[tid] += rz[tid+s]; rx[tid] += rx[tid+s];
            ry[tid] += ry[tid+s]; rzz[tid] += rzz[tid+s];
        }
        __syncthreads();
    }
    if (tid == 0) {
        float* qb_ = q + (size_t)bat * 52;
        atomicAdd(qb_ + qb,      rz[0]);
        atomicAdd(qb_ + 16 + qb, rx[0]);
        atomicAdd(qb_ + 32 + qb, ry[0]);
        if (qb < 4) atomicAdd(qb_ + 48 + qb, rzz[0]);
    }
}

// ---------------- host-side helpers ----------------
static inline void cvt(const float* s, unsigned short* d, int rows, int cols, int colsPad,
                       int rowsPad, hipStream_t st) {
    int total = rowsPad * colsPad;
    cvt_pad_kernel<<<(total + 255) / 256, 256, 0, st>>>(s, d, rows, cols, colsPad, total);
}
static inline void gemm(const unsigned short* A, const unsigned short* B, float* C,
                        int N, int K, hipStream_t st) {
    int Npad = (N + 15) & ~15;
    dim3 g((Npad + 63) / 64, 8);
    gemm_wmma_kernel<<<g, 128, 0, st>>>(A, B, C, N, K);
}
static inline void ep(const float* C, const float* bias, const float* skip,
                      const float* g, const float* b, float* oF, unsigned short* oB,
                      int N, int ldb, int mode, hipStream_t st) {
    epilogue_kernel<<<128, 256, 0, st>>>(C, bias, skip, g, b, oF, oB, N, ldb, mode);
}

// input index map (setup_inputs dict order, params flattened in insertion order)
enum {
    IN_X = 0, IN_W1, IN_B1, IN_G1, IN_BE1, IN_W2, IN_B2, IN_G2, IN_BE2,
    IN_W3, IN_B3, IN_G3, IN_BE3, IN_WP1, IN_BP1, IN_WP2, IN_BP2, IN_QW,
    IN_WQ1, IN_BQ1, IN_GQ1, IN_BEQ1, IN_WQ2, IN_BQ2, IN_GQ2, IN_BEQ2,
    IN_WO1, IN_BO1, IN_WO2, IN_BO2, IN_WO3, IN_BO3, IN_WS1, IN_BS1, IN_WS2, IN_BS2
};

extern "C" void kernel_launch(void* const* d_in, const int* in_sizes, int n_in,
                              void* d_out, int out_size, void* d_ws, size_t ws_size,
                              hipStream_t stream) {
    (void)in_sizes; (void)n_in; (void)out_size; (void)ws_size;
    char* W = (char*)d_ws;
    float2*         state = (float2*)(W + OFF_STATE);
    float*          encg  = (float*)(W + OFF_ENCG);
    float*          layg  = (float*)(W + OFF_LAYG);
    float*          Cbuf  = (float*)(W + OFF_C);
    float*          skip1 = (float*)(W + OFF_SK1);
    float*          skip2 = (float*)(W + OFF_SK2);
    float*          xpre  = (float*)(W + OFF_XPRE);
    float*          qout  = (float*)(W + OFF_QOUT);
    unsigned short* xs   = (unsigned short*)(W + OFF_XS);
    unsigned short* w1s  = (unsigned short*)(W + OFF_W1S);
    unsigned short* w2s  = (unsigned short*)(W + OFF_W2S);
    unsigned short* w3s  = (unsigned short*)(W + OFF_W3S);
    unsigned short* wp1s = (unsigned short*)(W + OFF_WP1S);
    unsigned short* wp2s = (unsigned short*)(W + OFF_WP2S);
    unsigned short* wq1s = (unsigned short*)(W + OFF_WQ1S);
    unsigned short* wq2s = (unsigned short*)(W + OFF_WQ2S);
    unsigned short* wo1s = (unsigned short*)(W + OFF_WO1S);
    unsigned short* wo2s = (unsigned short*)(W + OFF_WO2S);
    unsigned short* wo3s = (unsigned short*)(W + OFF_WO3S);
    unsigned short* ws1s = (unsigned short*)(W + OFF_WS1S);
    unsigned short* ws2s = (unsigned short*)(W + OFF_WS2S);
    unsigned short* a1s  = (unsigned short*)(W + OFF_A1S);
    unsigned short* a2s  = (unsigned short*)(W + OFF_A2S);
    unsigned short* a3s  = (unsigned short*)(W + OFF_A3S);
    unsigned short* ats  = (unsigned short*)(W + OFF_ATS);
    unsigned short* aqs  = (unsigned short*)(W + OFF_AQS);
    unsigned short* ap1s = (unsigned short*)(W + OFF_AP1S);
    unsigned short* ap2s = (unsigned short*)(W + OFF_AP2S);
    unsigned short* ah1s = (unsigned short*)(W + OFF_AH1S);
    unsigned short* ah2s = (unsigned short*)(W + OFF_AH2S);

    const float* x   = (const float*)d_in[IN_X];
    float*       out = (float*)d_out;

    // zero the staging arena + q accumulator
    {
        int nf = (int)((OFF_END - OFF_BF) / 4);
        zero_f32_kernel<<<(nf + 255) / 256, 256, 0, stream>>>((float*)(W + OFF_BF), nf);
        zero_f32_kernel<<<(128 * 52 + 255) / 256, 256, 0, stream>>>(qout, 128 * 52);
    }

    // ---- stage 0: bf16 staging (zero-padded) ----
    cvt(x,                          xs,   128, 512, 512, 128, stream);
    cvt((const float*)d_in[IN_W1],  w1s,  512, 512, 512, 512, stream);
    cvt((const float*)d_in[IN_W2],  w2s,  256, 512, 512, 256, stream);
    cvt((const float*)d_in[IN_W3],  w3s,  128, 256, 256, 128, stream);
    cvt((const float*)d_in[IN_WP1], wp1s, 64, 128, 128, 64, stream);
    cvt((const float*)d_in[IN_WP2], wp2s, 16, 64, 64, 16, stream);
    cvt((const float*)d_in[IN_WQ1], wq1s, 256, 52, 64, 256, stream);
    cvt((const float*)d_in[IN_WQ2], wq2s, 128, 256, 256, 128, stream);
    cvt((const float*)d_in[IN_WO1], wo1s, 64, 128, 128, 64, stream);
    cvt((const float*)d_in[IN_WO2], wo2s, 32, 64, 64, 32, stream);
    cvt((const float*)d_in[IN_WO3], wo3s, 1, 32, 32, 16, stream);
    cvt((const float*)d_in[IN_WS1], ws1s, 256, 512, 512, 256, stream);
    cvt((const float*)d_in[IN_WS2], ws2s, 128, 512, 512, 128, stream);

    // ---- stage 1: classical front ----
    gemm(xs, w1s, Cbuf, 512, 512, stream);
    ep(Cbuf, (const float*)d_in[IN_B1], nullptr, (const float*)d_in[IN_G1],
       (const float*)d_in[IN_BE1], nullptr, a1s, 512, 512, 1, stream);          // x1
    gemm(xs, ws1s, Cbuf, 256, 512, stream);
    ep(Cbuf, (const float*)d_in[IN_BS1], nullptr, nullptr, nullptr,
       skip1, nullptr, 256, 0, 0, stream);                                       // skip1
    gemm(xs, ws2s, Cbuf, 128, 512, stream);
    ep(Cbuf, (const float*)d_in[IN_BS2], nullptr, nullptr, nullptr,
       skip2, nullptr, 128, 0, 0, stream);                                       // skip2
    gemm(a1s, w2s, Cbuf, 256, 512, stream);
    ep(Cbuf, (const float*)d_in[IN_B2], skip1, (const float*)d_in[IN_G2],
       (const float*)d_in[IN_BE2], nullptr, a2s, 256, 256, 1, stream);          // x2
    gemm(a2s, w3s, Cbuf, 128, 256, stream);
    ep(Cbuf, (const float*)d_in[IN_B3], skip2, (const float*)d_in[IN_G3],
       (const float*)d_in[IN_BE3], nullptr, a3s, 128, 128, 1, stream);          // x3
    gemm(a3s, wp1s, Cbuf, 64, 128, stream);
    ep(Cbuf, (const float*)d_in[IN_BP1], nullptr, nullptr, nullptr,
       nullptr, ats, 64, 64, 2, stream);                                         // t
    gemm(ats, wp2s, Cbuf, 16, 64, stream);
    ep(Cbuf, (const float*)d_in[IN_BP2], nullptr, nullptr, nullptr,
       xpre, nullptr, 16, 0, 3, stream);                                         // x_pre

    // ---- stage 2: quantum simulation ----
    prep_enc_kernel<<<(NBATCH * NQ + 255) / 256, 256, 0, stream>>>(xpre, encg);
    prep_layer_kernel<<<1, 128, 0, stream>>>((const float*)d_in[IN_QW], layg);
    q_init_kernel<<<32768, 256, 0, stream>>>(state);

    dim3 gLow(16, NBATCH), gHi(128, NBATCH), gCn(64, NBATCH), gCz(256, NBATCH);
    // encoding (RY*RZ*H fused per qubit)
    q_apply_low_kernel<<<gLow, 256, 0, stream>>>(state, encg, NQ * 8);
    for (int qb = 0; qb < 4; ++qb)
        q_gate_hi_kernel<<<gHi, 256, 0, stream>>>(state, encg, NQ * 8, qb, 15 - qb);

    for (int l = 0; l < NLAY; ++l) {
        const float* gl = layg + (size_t)l * NQ * 8;
        q_apply_low_kernel<<<gLow, 256, 0, stream>>>(state, gl, 0);
        for (int qb = 0; qb < 4; ++qb)
            q_gate_hi_kernel<<<gHi, 256, 0, stream>>>(state, gl, 0, qb, 15 - qb);
        if (l % 3 == 0) {
            for (int i = 0; i < NQ - 1; ++i)
                q_cnot_kernel<<<gCn, 256, 0, stream>>>(state, 15 - i, 15 - (i + 1));
            q_cnot_kernel<<<gCn, 256, 0, stream>>>(state, 15 - 15, 15 - 0);
        } else if (l % 3 == 1) {
            for (int i = 0; i < NQ / 2; ++i)
                q_cnot_kernel<<<gCn, 256, 0, stream>>>(state, 15 - i, 15 - (i + 8));
        } else {
            for (int i = 1; i < NQ; ++i)
                q_cnot_kernel<<<gCn, 256, 0, stream>>>(state, 15 - 0, 15 - i);
        }
        if (l % 2 == 0)
            q_cz_layer_kernel<<<gCz, 256, 0, stream>>>(state);
    }
    for (int qb = 0; qb < NQ; ++qb)
        q_expect_kernel<<<dim3(32, NBATCH), 256, 0, stream>>>(state, qout, qb);

    // ---- stage 3: classical back ----
    cvt(qout, aqs, 128, 52, 64, 128, stream);
    gemm(aqs, wq1s, Cbuf, 256, 64, stream);
    ep(Cbuf, (const float*)d_in[IN_BQ1], nullptr, (const float*)d_in[IN_GQ1],
       (const float*)d_in[IN_BEQ1], nullptr, ap1s, 256, 256, 1, stream);        // p1
    gemm(ap1s, wq2s, Cbuf, 128, 256, stream);
    ep(Cbuf, (const float*)d_in[IN_BQ2], skip2, (const float*)d_in[IN_GQ2],
       (const float*)d_in[IN_BEQ2], nullptr, ap2s, 128, 128, 1, stream);        // p2
    gemm(ap2s, wo1s, Cbuf, 64, 128, stream);
    ep(Cbuf, (const float*)d_in[IN_BO1], nullptr, nullptr, nullptr,
       nullptr, ah1s, 64, 64, 2, stream);                                        // h (64)
    gemm(ah1s, wo2s, Cbuf, 32, 64, stream);
    ep(Cbuf, (const float*)d_in[IN_BO2], nullptr, nullptr, nullptr,
       nullptr, ah2s, 32, 32, 2, stream);                                        // h (32)
    gemm(ah2s, wo3s, Cbuf, 1, 32, stream);
    ep(Cbuf, (const float*)d_in[IN_BO3], nullptr, nullptr, nullptr,
       out, nullptr, 1, 0, 0, stream);                                           // output
}
